// TransformerBlockWithGroupedAttention_67791763800460
// MI455X (gfx1250) — matrix operand order repaired and checked by reading, not compiled, not run on hardware
//
#include <hip/hip_runtime.h>
#include <hip/hip_bf16.h>

// ---------------------------------------------------------------- types
typedef __bf16 bf16_t;
typedef __attribute__((ext_vector_type(16))) bf16_t v16bf;
typedef __attribute__((ext_vector_type(8)))  float  v8f;
typedef __attribute__((ext_vector_type(4)))  int    v4i;

union V16 { v16bf v; uint4 q[2]; };

// ---------------------------------------------------------------- sizes (fixed by reference)
constexpr int  Bb = 4, T = 1024, Cc = 2048, G = 8, H = 8, GS = 256, HD = 32;
constexpr int  BT = Bb * T;          // 4096 token rows
constexpr int  FF = 4 * Cc;          // 8192

// workspace layout (bytes)
constexpr size_t SZ_QKV1 = (size_t)Bb * G * H * T * HD * 2;   // 16 MiB per q/k/v
constexpr size_t SZ_ACT4 = (size_t)BT * Cc * 4;               // 32 MiB f32 act
constexpr size_t SZ_ACT2 = (size_t)BT * Cc * 2;               // 16 MiB bf16 act
constexpr size_t OFF_Q     = 0;
constexpr size_t OFF_K     = SZ_QKV1;
constexpr size_t OFF_V     = 2 * SZ_QKV1;
constexpr size_t OFF_S1    = 0;                         // reuses q/k/v (dead by then)
constexpr size_t OFF_S2    = 0;                         // reuses s1 (dead by then)
constexpr size_t OFF_CTXB  = 3 * SZ_QKV1;               //  50331648 (bf16 ctx)
constexpr size_t OFF_H     = OFF_CTXB + SZ_ACT2;        //  67108864 (f32 h)
constexpr size_t OFF_HB    = OFF_H + SZ_ACT4;           // 100663296 (bf16 h)
constexpr size_t OFF_U     = OFF_HB + SZ_ACT2;          // 117440512 (bf16 ffn hidden, 64 MiB)
constexpr size_t OFF_XB    = OFF_U + (size_t)BT * FF * 2;     // 184549376
constexpr size_t OFF_WQKVB = OFF_XB + SZ_ACT2;                // 201326592
constexpr size_t OFF_WOUTB = OFF_WQKVB + (size_t)G * 3 * GS * GS * 2;  // 204472320
constexpr size_t OFF_W1B   = OFF_WOUTB + (size_t)G * GS * GS * 2;      // 205520896
constexpr size_t OFF_W2B   = OFF_W1B + (size_t)FF * Cc * 2;            // 239075328

// ---------------------------------------------------------------- helpers
__device__ __forceinline__ unsigned short bf_bits(float f) {
  union { float f; unsigned u; } c; c.f = f;
  unsigned u = c.u + 0x7FFFu + ((c.u >> 16) & 1u);   // round-to-nearest-even
  return (unsigned short)(u >> 16);
}
__device__ __forceinline__ bf16_t f2bf(float f) {
  union { unsigned short s; bf16_t b; } c; c.s = bf_bits(f); return c.b;
}
__device__ __forceinline__ unsigned bfpair(float lo, float hi) {
  return (unsigned)bf_bits(lo) | ((unsigned)bf_bits(hi) << 16);
}

// ---- CDNA5 async global->LDS copy (VGPR-bypassing, tracked by ASYNCcnt) ----
#if defined(__gfx1250__) && __has_builtin(__builtin_amdgcn_global_load_async_to_lds_b128)
#define USE_ASYNC_LDS 1
typedef __attribute__((address_space(1))) v4i glb_v4i;   // global b128 pointee
typedef __attribute__((address_space(3))) v4i lds_v4i;   // LDS b128 pointee
#endif

__device__ __forceinline__ void async_b128(const void* g, void* l) {
#ifdef USE_ASYNC_LDS
  __builtin_amdgcn_global_load_async_to_lds_b128((glb_v4i*)g, (lds_v4i*)l, 0, 0);
#else
  *(uint4*)l = *(const uint4*)g;
#endif
}
__device__ __forceinline__ void async_wait0() {
#ifdef USE_ASYNC_LDS
#if __has_builtin(__builtin_amdgcn_s_wait_asynccnt)
  __builtin_amdgcn_s_wait_asynccnt(0);
#else
  asm volatile("s_wait_asynccnt 0x0" ::: "memory");
#endif
#endif
}

// ================================================================ f32 -> bf16 bulk convert
__global__ __launch_bounds__(256) void cvt_kernel(const float* __restrict__ in,
                                                  bf16_t* __restrict__ out, long n) {
  const long i = ((long)blockIdx.x * 256 + threadIdx.x) * 8;
  if (i + 8 <= n) {
    float4 a = *(const float4*)(in + i);
    float4 b = *(const float4*)(in + i + 4);
    *(uint4*)(out + i) = make_uint4(bfpair(a.x, a.y), bfpair(a.z, a.w),
                                    bfpair(b.x, b.y), bfpair(b.z, b.w));
  }
}

// ================================================================ GEMM
// out = A(MxK,bf16) @ W(NxK,bf16)^T + bias, fused epilogues.
// MODE 1: gelu -> bf16 out        (FFN1)
// MODE 2: + resid -> f32 out      (out-proj + x ; FFN2 + h)
// MODE 3: qkv scatter -> bf16 q/k/v^T
struct GemmArgs {
  const bf16_t* A;   long lda; int  gAoff;    // per-group column offset into A
  const bf16_t* W;   long ldw; long gWoff;
  const float*  bias;          int  gBoff;
  float*        outF; long ldo; int gOoff;
  const float*  resid; long ldr; int gRoff;
  bf16_t*       outB;
  bf16_t*       q; bf16_t* kk; bf16_t* vv;
  int M, N, K;
};

template <int MODE>
__global__ __launch_bounds__(256) void gemm_kernel(GemmArgs ga) {
  // double-buffered bf16 tiles; 40-elem stride keeps 16B alignment, avoids conflicts
  __shared__ __align__(16) bf16_t As[2][128][40];
  __shared__ __align__(16) bf16_t Bs[2][128][40];

  const int tid   = threadIdx.x;
  const int lid   = tid & 31;
  const int wave  = tid >> 5;
  const int g     = blockIdx.z;
  const long rowBlock = (long)blockIdx.x * 128;
  const int  colBlock = blockIdx.y * 128;

  const bf16_t* Ab   = ga.A + (long)g * ga.gAoff;
  const bf16_t* Wb   = ga.W + (long)g * ga.gWoff;
  const float*  bias = ga.bias + (long)g * ga.gBoff;

  const int lr = tid >> 1;          // tile row this thread stages
  const int lc = (tid & 1) * 16;    // tile col base (16 bf16 per thread per array)

  const int mBase = (wave >> 1) * 32;
  const int nBase = (wave & 1) * 64;
  const int mlane = lid & 15;
  const int hi    = lid >> 4;
  const int kA    = hi * 8;         // A-frag: K = {kA..kA+7, kA+16..kA+23}
  const int kB    = hi * 8;         // B-frag: K = {kB..kB+7, kB+16..kB+23}

  v8f acc[2][4] = {};

  auto issue = [&](int k0, int buf) {
    const bf16_t* asrc = Ab + (rowBlock + lr) * ga.lda + k0 + lc;
    async_b128(asrc,     &As[buf][lr][lc]);
    async_b128(asrc + 8, &As[buf][lr][lc + 8]);
    const bf16_t* bsrc = Wb + (long)(colBlock + lr) * ga.ldw + k0 + lc;
    async_b128(bsrc,     &Bs[buf][lr][lc]);
    async_b128(bsrc + 8, &Bs[buf][lr][lc + 8]);
  };

  const int nk = ga.K >> 5;
  issue(0, 0);
  for (int kt = 0; kt < nk; ++kt) {
    async_wait0();        // this wave's async fills (tiles <= kt) have landed in LDS
    __syncthreads();      // all waves' fills landed; all waves done reading buf^1
    if (kt + 1 < nk) issue((kt + 1) << 5, (kt + 1) & 1);   // prefetch next tile
    const int buf = kt & 1;

    v16bf afrag[2], bfrag[4];
#pragma unroll
    for (int mt = 0; mt < 2; ++mt) {
      V16 t;
      const bf16_t* p = &As[buf][mBase + mt * 16 + mlane][0];
      t.q[0] = *(const uint4*)(p + kA);
      t.q[1] = *(const uint4*)(p + kA + 16);
      afrag[mt] = t.v;
    }
#pragma unroll
    for (int nt = 0; nt < 4; ++nt) {
      V16 t;
      const bf16_t* p = &Bs[buf][nBase + nt * 16 + mlane][0];
      t.q[0] = *(const uint4*)(p + kB);
      t.q[1] = *(const uint4*)(p + kB + 16);
      bfrag[nt] = t.v;
    }
#pragma unroll
    for (int mt = 0; mt < 2; ++mt)
#pragma unroll
      for (int nt = 0; nt < 4; ++nt)
        acc[mt][nt] = __builtin_amdgcn_wmma_f32_16x16x32_bf16(
            false, afrag[mt], false, bfrag[nt], (short)0, acc[mt][nt], false, false);
  }

  // ---- epilogue (C/D layout: lanes 0-15 -> M=e, lanes 16-31 -> M=e+8; N=lane%16)
#pragma unroll
  for (int mt = 0; mt < 2; ++mt) {
#pragma unroll
    for (int nt = 0; nt < 4; ++nt) {
#pragma unroll
      for (int e = 0; e < 8; ++e) {
        const long row = rowBlock + mBase + mt * 16 + e + hi * 8;
        const int  col = colBlock + nBase + nt * 16 + mlane;
        float val = acc[mt][nt][e] + bias[col];
        if (MODE == 1) {
          float gl = 0.5f * val * (1.0f + erff(val * 0.70710678118654752f));
          ga.outB[row * ga.ldo + col] = f2bf(gl);
        } else if (MODE == 2) {
          val += ga.resid[row * ga.ldr + (long)g * ga.gRoff + col];
          ga.outF[row * ga.ldo + (long)g * ga.gOoff + col] = val;
        } else if (MODE == 3) {
          const int b   = (int)(row >> 10);
          const int t   = (int)(row & 1023);
          const int sec = col >> 8;
          const int idx = col & 255;
          const int hh  = idx >> 5;
          const int d   = idx & 31;
          const long bgh = ((long)b * G + g) * H + hh;
          const bf16_t bv = f2bf(val);
          if (sec == 0)      ga.q [(bgh * T + t) * HD + d] = bv;   // (B,G,H,T,hd)
          else if (sec == 1) ga.kk[(bgh * T + t) * HD + d] = bv;   // (B,G,H,T,hd)
          else               ga.vv[(bgh * HD + d) * T + t] = bv;   // (B,G,H,hd,T)  V^T
        }
      }
    }
  }
}

// ================================================================ flash attention
// grid (T/64, B*G*H), 128 threads (4 waves). Wave w handles 16 query rows.
__global__ __launch_bounds__(128) void attn_kernel(
    const bf16_t* __restrict__ q, const bf16_t* __restrict__ k,
    const bf16_t* __restrict__ vt, bf16_t* __restrict__ ctxb) {
  __shared__ __align__(16) bf16_t pls[4][16][40];   // per-wave probs transpose staging

  const int tid   = threadIdx.x;
  const int lid   = tid & 31;
  const int wave  = tid >> 5;
  const int mlane = lid & 15;
  const int hi    = lid >> 4;
  const int bgh   = blockIdx.y;
  const int qr0   = blockIdx.x * 64 + wave * 16;

  // Q fragment (A layout), held in registers for the whole key loop
  const bf16_t* qb = q + ((long)bgh * T + qr0 + mlane) * HD;
  V16 tq;
  tq.q[0] = *(const uint4*)(qb + hi * 8);
  tq.q[1] = *(const uint4*)(qb + hi * 8 + 16);
  const v16bf aq = tq.v;

  const bf16_t* kb = k  + (long)bgh * T * HD;
  const bf16_t* vb = vt + (long)bgh * HD * T;

  float mrow[8], lrow[8];
  v8f cacc[2] = {};
#pragma unroll
  for (int e = 0; e < 8; ++e) { mrow[e] = -1e30f; lrow[e] = 0.f; }

  const float scale = 0.17677669529663688f;   // 1/sqrt(32)
  const int   nkb   = (qr0 + 16 + 31) >> 5;   // causal: keys 0 .. qr0+15

  for (int kblk = 0; kblk < nkb; ++kblk) {
    const int key0 = kblk * 32;

    // ---- scores: one WMMA per 16x16 tile (K = hd = 32)
    v8f s[2];
#pragma unroll
    for (int kt = 0; kt < 2; ++kt) {
      const bf16_t* kr = kb + (long)(key0 + kt * 16 + mlane) * HD + hi * 8;
      V16 tk;
      tk.q[0] = *(const uint4*)(kr);
      tk.q[1] = *(const uint4*)(kr + 16);
      v8f z = {};
      s[kt] = __builtin_amdgcn_wmma_f32_16x16x32_bf16(
          false, aq, false, tk.v, (short)0, z, false, false);
    }

    // ---- online softmax update (rows split: lanes 0-15 -> e, 16-31 -> e+8)
    float alpha[8];
#pragma unroll
    for (int e = 0; e < 8; ++e) {
      const int row = qr0 + e + hi * 8;
      const int c0  = key0 + mlane;
      float s0 = s[0][e] * scale;
      float s1 = s[1][e] * scale;
      if (c0 > row)      s0 = -1e30f;          // causal mask
      if (c0 + 16 > row) s1 = -1e30f;
      float mx = fmaxf(s0, s1);
      mx = fmaxf(mx, __shfl_xor(mx, 1, 32));
      mx = fmaxf(mx, __shfl_xor(mx, 2, 32));
      mx = fmaxf(mx, __shfl_xor(mx, 4, 32));
      mx = fmaxf(mx, __shfl_xor(mx, 8, 32));
      const float mn = fmaxf(mrow[e], mx);
      alpha[e] = __expf(mrow[e] - mn);
      mrow[e]  = mn;
      const float p0 = __expf(s0 - mn);
      const float p1 = __expf(s1 - mn);
      s[0][e] = p0; s[1][e] = p1;
      float sum = p0 + p1;
      sum += __shfl_xor(sum, 1, 32);
      sum += __shfl_xor(sum, 2, 32);
      sum += __shfl_xor(sum, 4, 32);
      sum += __shfl_xor(sum, 8, 32);
      lrow[e] = lrow[e] * alpha[e] + sum;
      cacc[0][e] *= alpha[e];
      cacc[1][e] *= alpha[e];
    }

    // ---- transpose probs through per-wave LDS (D layout -> A layout)
#pragma unroll
    for (int kt = 0; kt < 2; ++kt)
#pragma unroll
      for (int e = 0; e < 8; ++e)
        pls[wave][e + hi * 8][kt * 16 + mlane] = f2bf(s[kt][e]);
    asm volatile("s_wait_dscnt 0x0" ::: "memory");   // per-wave LDS RAW fence

    V16 tp;
    const bf16_t* pr = &pls[wave][mlane][0];
    tp.q[0] = *(const uint4*)(pr + hi * 8);
    tp.q[1] = *(const uint4*)(pr + hi * 8 + 16);

    // ---- ctx += P @ V  (two WMMAs: hd split 16+16; V^T rows are contiguous keys)
#pragma unroll
    for (int nt = 0; nt < 2; ++nt) {
      const bf16_t* vr = vb + (long)(nt * 16 + mlane) * T + key0 + hi * 8;
      V16 tv;
      tv.q[0] = *(const uint4*)(vr);
      tv.q[1] = *(const uint4*)(vr + 16);
      cacc[nt] = __builtin_amdgcn_wmma_f32_16x16x32_bf16(
          false, tp.v, false, tv.v, (short)0, cacc[nt], false, false);
    }
  }

  // ---- normalize + store ctx as (B,T,C) bf16 (A operand of the out-proj GEMM)
  const int b = bgh >> 6;
  const int g = (bgh >> 3) & 7;
  const int h = bgh & 7;
#pragma unroll
  for (int nt = 0; nt < 2; ++nt)
#pragma unroll
    for (int e = 0; e < 8; ++e) {
      const int row = qr0 + e + hi * 8;
      ctxb[((long)b * T + row) * Cc + g * GS + h * HD + nt * 16 + mlane] =
          f2bf(cacc[nt][e] / lrow[e]);
    }
}

// ================================================================ LayerNorm (f32 out + optional bf16 copy)
__global__ __launch_bounds__(256) void ln_kernel(const float* __restrict__ in,
                                                 const float* __restrict__ gm,
                                                 const float* __restrict__ bt,
                                                 float* __restrict__ out,
                                                 bf16_t* __restrict__ ob) {
  __shared__ float sb[256], sb2[256];
  const int row = blockIdx.x;
  const int tid = threadIdx.x;
  const float* x = in + (long)row * Cc;
  float s = 0.f, s2 = 0.f;
  for (int i = tid; i < Cc; i += 256) { const float v = x[i]; s += v; s2 += v * v; }
  sb[tid] = s; sb2[tid] = s2;
  __syncthreads();
  for (int off = 128; off > 0; off >>= 1) {
    if (tid < off) { sb[tid] += sb[tid + off]; sb2[tid] += sb2[tid + off]; }
    __syncthreads();
  }
  const float mean = sb[0] * (1.f / Cc);
  const float var  = sb2[0] * (1.f / Cc) - mean * mean;
  const float rinv = rsqrtf(var + 1e-5f);
  float* o = out + (long)row * Cc;
  for (int i = tid; i < Cc; i += 256) {
    const float v = (x[i] - mean) * rinv * gm[i] + bt[i];
    o[i] = v;
    if (ob) ob[(long)row * Cc + i] = f2bf(v);
  }
}

// ================================================================ host
extern "C" void kernel_launch(void* const* d_in, const int* in_sizes, int n_in,
                              void* d_out, int out_size, void* d_ws, size_t ws_size,
                              hipStream_t stream) {
  (void)in_sizes; (void)n_in; (void)out_size; (void)ws_size;

  const float* x     = (const float*)d_in[0];
  const float* Wqkv  = (const float*)d_in[1];
  const float* bqkv  = (const float*)d_in[2];
  const float* Wout  = (const float*)d_in[3];
  const float* bout  = (const float*)d_in[4];
  const float* ln1g  = (const float*)d_in[5];
  const float* ln1b  = (const float*)d_in[6];
  const float* W1    = (const float*)d_in[7];
  const float* b1    = (const float*)d_in[8];
  const float* W2    = (const float*)d_in[9];
  const float* b2    = (const float*)d_in[10];
  const float* ln2g  = (const float*)d_in[11];
  const float* ln2b  = (const float*)d_in[12];

  char* ws = (char*)d_ws;
  bf16_t* q     = (bf16_t*)(ws + OFF_Q);
  bf16_t* kk    = (bf16_t*)(ws + OFF_K);
  bf16_t* vv    = (bf16_t*)(ws + OFF_V);
  float*  s1    = (float*) (ws + OFF_S1);
  float*  s2    = (float*) (ws + OFF_S2);
  bf16_t* ctxb  = (bf16_t*)(ws + OFF_CTXB);
  float*  h     = (float*) (ws + OFF_H);
  bf16_t* hb    = (bf16_t*)(ws + OFF_HB);
  bf16_t* u     = (bf16_t*)(ws + OFF_U);
  bf16_t* xb    = (bf16_t*)(ws + OFF_XB);
  bf16_t* wqkvb = (bf16_t*)(ws + OFF_WQKVB);
  bf16_t* woutb = (bf16_t*)(ws + OFF_WOUTB);
  bf16_t* w1b   = (bf16_t*)(ws + OFF_W1B);
  bf16_t* w2b   = (bf16_t*)(ws + OFF_W2B);

  // 0) one-shot f32 -> bf16 conversions (bandwidth-bound; weights then stay hot in L2)
  cvt_kernel<<<(BT * (long)Cc) / 2048, 256, 0, stream>>>(x, xb, (long)BT * Cc);
  cvt_kernel<<<((long)G * 3 * GS * GS) / 2048, 256, 0, stream>>>(Wqkv, wqkvb, (long)G * 3 * GS * GS);
  cvt_kernel<<<((long)G * GS * GS) / 2048, 256, 0, stream>>>(Wout, woutb, (long)G * GS * GS);
  cvt_kernel<<<((long)FF * Cc) / 2048, 256, 0, stream>>>(W1, w1b, (long)FF * Cc);
  cvt_kernel<<<((long)Cc * FF) / 2048, 256, 0, stream>>>(W2, w2b, (long)Cc * FF);

  // 1) grouped QKV projection -> q,k (B,G,H,T,hd), v transposed (B,G,H,hd,T), bf16
  {
    GemmArgs a{};
    a.A = xb;    a.lda = Cc; a.gAoff = GS;
    a.W = wqkvb; a.ldw = GS; a.gWoff = (long)3 * GS * GS;
    a.bias = bqkv; a.gBoff = 3 * GS;
    a.q = q; a.kk = kk; a.vv = vv;
    a.M = BT; a.N = 3 * GS; a.K = GS;
    gemm_kernel<3><<<dim3(BT / 128, (3 * GS) / 128, G), 256, 0, stream>>>(a);
  }

  // 2) causal flash attention -> ctxb (B,T,C) bf16
  attn_kernel<<<dim3(T / 64, Bb * G * H), 128, 0, stream>>>(q, kk, vv, ctxb);

  // 3) grouped out-projection + residual(x) -> s1 (B,T,C) f32
  {
    GemmArgs a{};
    a.A = ctxb;  a.lda = Cc; a.gAoff = GS;
    a.W = woutb; a.ldw = GS; a.gWoff = (long)GS * GS;
    a.bias = bout; a.gBoff = GS;
    a.outF = s1; a.ldo = Cc; a.gOoff = GS;
    a.resid = x; a.ldr = Cc; a.gRoff = GS;
    a.M = BT; a.N = GS; a.K = GS;
    gemm_kernel<2><<<dim3(BT / 128, GS / 128, G), 256, 0, stream>>>(a);
  }

  // 4) LN1 -> h (f32) + hb (bf16)
  ln_kernel<<<BT, 256, 0, stream>>>(s1, ln1g, ln1b, h, hb);

  // 5) FFN1: gelu(h @ W1^T + b1) -> u (bf16)
  {
    GemmArgs a{};
    a.A = hb;  a.lda = Cc; a.gAoff = 0;
    a.W = w1b; a.ldw = Cc; a.gWoff = 0;
    a.bias = b1; a.gBoff = 0;
    a.outB = u; a.ldo = FF;
    a.M = BT; a.N = FF; a.K = Cc;
    gemm_kernel<1><<<dim3(BT / 128, FF / 128, 1), 256, 0, stream>>>(a);
  }

  // 6) FFN2: u @ W2^T + b2 + h -> s2
  {
    GemmArgs a{};
    a.A = u;   a.lda = FF; a.gAoff = 0;
    a.W = w2b; a.ldw = FF; a.gWoff = 0;
    a.bias = b2; a.gBoff = 0;
    a.outF = s2; a.ldo = Cc; a.gOoff = 0;
    a.resid = h; a.ldr = Cc; a.gRoff = 0;
    a.M = BT; a.N = Cc; a.K = FF;
    gemm_kernel<2><<<dim3(BT / 128, Cc / 128, 1), 256, 0, stream>>>(a);
  }

  // 7) LN2 -> output
  ln_kernel<<<BT, 256, 0, stream>>>(s2, ln2g, ln2b, (float*)d_out, nullptr);
}
